// MagicPoint_87505663689167
// MI455X (gfx1250) — matrix operand order repaired and checked by reading, not compiled, not run on hardware
//
#include <hip/hip_runtime.h>

typedef __attribute__((ext_vector_type(16))) _Float16 v16h;
typedef __attribute__((ext_vector_type(8)))  _Float16 v8h;
typedef __attribute__((ext_vector_type(8)))  float    v8f;

union UH { v16h v; v8h h[2]; };

// ---------------------------------------------------------------------------
// Weight reformat: w (cout,cin,ks,ks) f32  ->  B[k][coPad] f16,
// k = (ky*ks+kx)*cin + ci   (spatial-major, channel-minor)
// ---------------------------------------------------------------------------
__global__ void reformat_w(const float* __restrict__ w, _Float16* __restrict__ dst,
                           int cin, int cout, int coutPad, int ks) {
    int K = ks * ks * cin;
    int total = K * coutPad;
    int id = blockIdx.x * blockDim.x + threadIdx.x;
    if (id >= total) return;
    int kk = id / coutPad, co = id % coutPad;
    float v = 0.f;
    if (co < cout) {
        int sp = kk / cin, ci = kk % cin;
        int ky = sp / ks, kx = sp % ks;
        v = w[(((size_t)co * cin + ci) * ks + ky) * ks + kx];
    }
    dst[id] = (_Float16)v;
}

// ---------------------------------------------------------------------------
// conv0: 1->64 ch, 3x3 pad1, input NCHW f32 (cin=1), output NHWC f16 + ReLU
// ---------------------------------------------------------------------------
__global__ void conv0_direct(const float* __restrict__ x, const float* __restrict__ w,
                             const float* __restrict__ b, _Float16* __restrict__ out) {
    const int H = 240, W = 320, C = 64;
    int id = blockIdx.x * blockDim.x + threadIdx.x;
    int total = 8 * H * W * C;
    if (id >= total) return;
    int co = id % C; int pos = id / C;
    int xx = pos % W; int yy = (pos / W) % H; int n = pos / (W * H);
    float s = b[co];
    for (int ky = 0; ky < 3; ky++) {
        int iy = yy + ky - 1; if (iy < 0 || iy >= H) continue;
        for (int kx = 0; kx < 3; kx++) {
            int ix = xx + kx - 1; if (ix < 0 || ix >= W) continue;
            s += x[((size_t)n * H + iy) * W + ix] * w[co * 9 + ky * 3 + kx];
        }
    }
    out[id] = (_Float16)fmaxf(s, 0.f);
}

// ---------------------------------------------------------------------------
// Implicit-GEMM conv via v_wmma_f32_16x16x32_f16, N-blocked 4 tiles/wave.
// act: NHWC f16, Bw: [K][coutPad] f16, out NHWC f16 (or f32 for the head).
// One wave = one 16(M) x 64(N) output block; K looped in 32-steps with the
// A fragment (padding-checked gather) reused across the 4 N-tiles.
// Requires cin % 32 == 0.
// ---------------------------------------------------------------------------
__global__ __launch_bounds__(256)
void conv_wmma(const _Float16* __restrict__ act, const _Float16* __restrict__ Bw,
               const float* __restrict__ bias, _Float16* __restrict__ out16,
               float* __restrict__ out32,
               int Nimg, int H, int W, int cin, int cout, int coutPad,
               int ks, int doRelu) {
    const int M = Nimg * H * W;
    const int Mtiles = (M + 15) >> 4;
    const int Ntiles = coutPad >> 4;
    const int Ngroups = (Ntiles + 3) >> 2;
    int wid = blockIdx.x * (blockDim.x >> 5) + (threadIdx.x >> 5);
    if (wid >= Mtiles * Ngroups) return;          // wave-uniform exit
    int mtile = wid / Ngroups, ng = wid % Ngroups;
    int nt0 = ng << 2;
    int nb = Ntiles - nt0; if (nb > 4) nb = 4;    // wave-uniform

    int lane = threadIdx.x & 31;
    int half = lane >> 4, lr = lane & 15;

    int m = mtile * 16 + lr;                      // A row owned by this lane
    bool vm = (m < M);
    int mm = vm ? m : 0;
    int xx = mm % W, yy = (mm / W) % H, n = mm / (W * H);
    int pad = ks >> 1;
    int cpk = cin >> 5;                           // 32-channel chunks per (ky,kx)
    int Ksteps = ks * ks * cpk;

    v8f acc[4] = {{}, {}, {}, {}};
    for (int s = 0; s < Ksteps; s++) {
        int sp = s / cpk;
        int ci0 = (s % cpk) << 5;
        int ky = sp / ks, kx = sp % ks;           // ks==1 -> (0,0)
        int iy = yy + ky - pad, ix = xx + kx - pad;
        bool v = vm && iy >= 0 && iy < H && ix >= 0 && ix < W;
        v8h a0 = {}, a1 = {};
        if (v) {
            const v8h* ap = (const v8h*)(act + (((size_t)n * H + iy) * W + ix) * cin
                                             + ci0 + half * 8);
            a0 = ap[0];                           // K = ci0 + half*8 + 0..7
            a1 = ap[2];                           // K = ci0 + half*8 + 16..23
        }
        UH A; A.h[0] = a0; A.h[1] = a1;
        const _Float16* brow = Bw + ((size_t)(s * 32)) * coutPad + nt0 * 16 + lr;
#pragma unroll
        for (int j = 0; j < 4; j++) {
            if (j < nb) {
                UH B;
                for (int e = 0; e < 16; e++) {
                    int kq = (e < 8) ? (half * 8 + e) : (16 + half * 8 + (e - 8));
                    B.v[e] = brow[(size_t)kq * coutPad + j * 16];
                }
                acc[j] = __builtin_amdgcn_wmma_f32_16x16x32_f16(
                    false, A.v, false, B.v, (short)0, acc[j], false, false);
            }
        }
    }

#pragma unroll
    for (int j = 0; j < 4; j++) {
        if (j < nb) {
            int nco = (nt0 + j) * 16 + lr;        // D: lane owns column nco
            float bi = (nco < cout) ? bias[nco] : 0.f;
            for (int r = 0; r < 8; r++) {
                int mo = mtile * 16 + r + half * 8;  // D: vgpr r -> row r (+8 hi half)
                if (mo < M && nco < cout) {
                    float vv = acc[j][r] + bi;
                    if (doRelu) vv = fmaxf(vv, 0.f);
                    if (out32) out32[(size_t)mo * cout + nco] = vv;
                    else       out16[(size_t)mo * cout + nco] = (_Float16)vv;
                }
            }
        }
    }
}

// ---------------------------------------------------------------------------
// 2x2 max pool, stride 2, NHWC f16
// ---------------------------------------------------------------------------
__global__ void maxpool2(const _Float16* __restrict__ in, _Float16* __restrict__ out,
                         int Nimg, int H, int W, int C) {
    int Ho = H >> 1, Wo = W >> 1;
    int total = Nimg * Ho * Wo * C;
    int id = blockIdx.x * blockDim.x + threadIdx.x;
    if (id >= total) return;
    int c = id % C; int t = id / C;
    int xo = t % Wo; int yo = (t / Wo) % Ho; int n = t / (Wo * Ho);
    const _Float16* p0 = in + (((size_t)n * H + 2 * yo) * W + 2 * xo) * C + c;
    float a = (float)p0[0], b = (float)p0[C];
    float c2 = (float)p0[(size_t)W * C], d = (float)p0[(size_t)W * C + C];
    out[id] = (_Float16)fmaxf(fmaxf(a, b), fmaxf(c2, d));
}

// ---------------------------------------------------------------------------
// softmax over 65 ch (drop dustbin) + pixel shuffle(8): semi[9600][65] f32
// -> prob (8,240,320) f32 written directly into d_out section 0
// ---------------------------------------------------------------------------
__global__ void softmax_shuffle(const float* __restrict__ semi, float* __restrict__ prob) {
    int m = blockIdx.x * blockDim.x + threadIdx.x;
    if (m >= 9600) return;
    const float* s = semi + (size_t)m * 65;
    float mx = s[0];
    for (int c = 1; c < 65; c++) mx = fmaxf(mx, s[c]);
    float sum = 0.f;
    for (int c = 0; c < 65; c++) sum += expf(s[c] - mx);
    float inv = 1.f / sum;
    int wc = m % 40, hc = (m / 40) % 30, n = m / 1200;
    for (int c = 0; c < 64; c++) {
        int gy = c >> 3, gx = c & 7;
        prob[((size_t)n * 240 + hc * 8 + gy) * 320 + wc * 8 + gx] = expf(s[c] - mx) * inv;
    }
}

// ---------------------------------------------------------------------------
// Per-image box NMS. One 1024-thread block per image.
// Radix-select exact 1024th-largest cut -> compact -> bitonic sort (desc value,
// asc index) -> greedy suppression -> keep top 300 -> scatter.
// ---------------------------------------------------------------------------
#define HWPX 76800
__global__ __launch_bounds__(1024)
void nms_kernel(const float* __restrict__ prob, float* __restrict__ pnms,
                float* __restrict__ pred) {
    __shared__ int hist[256];
    __shared__ unsigned pref_s;
    __shared__ int need_s;
    __shared__ int cnt_s;
    __shared__ unsigned long long keys[2048];
    __shared__ float val_s[1024];
    __shared__ int   idx_s[1024];
    __shared__ float ys_s[1024], xs_s[1024];
    __shared__ int   keep_s[1024];

    int tid = threadIdx.x;
    const float* p = prob + (size_t)blockIdx.x * HWPX;
    float* pn = pnms + (size_t)blockIdx.x * HWPX;
    float* pr = pred + (size_t)blockIdx.x * HWPX;

    // ---- exact radix select of 1024th largest (all values are positive) ----
    unsigned pref = 0; int need = 1024;
    for (int shift = 24; shift >= 0; shift -= 8) {
        if (tid < 256) hist[tid] = 0;
        __syncthreads();
        unsigned maskHi = (shift == 24) ? 0u : (0xFFFFFFFFu << (shift + 8));
        for (int i = tid; i < HWPX; i += 1024) {
            unsigned u = __float_as_uint(p[i]);
            if ((u & maskHi) == (pref & maskHi))
                atomicAdd(&hist[(u >> shift) & 255], 1);
        }
        __syncthreads();
        if (tid == 0) {
            int cum = 0, v = 255;
            for (; v > 0; v--) {
                if (cum + hist[v] >= need) break;
                cum += hist[v];
            }
            pref_s = pref | ((unsigned)v << shift);
            need_s = need - cum;
        }
        __syncthreads();
        pref = pref_s; need = need_s;
        __syncthreads();
    }
    unsigned T = pref;

    // ---- compact candidates >= cut ----
    if (tid == 0) cnt_s = 0;
    __syncthreads();
    for (int i = tid; i < HWPX; i += 1024) {
        unsigned u = __float_as_uint(p[i]);
        if (u >= T) {
            int k = atomicAdd(&cnt_s, 1);
            if (k < 2048)
                keys[k] = ((unsigned long long)u << 32) | (unsigned)(0xFFFFFFFFu - (unsigned)i);
        }
    }
    __syncthreads();
    int ncand = cnt_s; if (ncand > 2048) ncand = 2048;
    for (int i = tid; i < 2048; i += 1024)
        if (i >= ncand) keys[i] = 0ull;
    __syncthreads();

    // ---- bitonic sort, descending (value desc, index asc) ----
    for (int k = 2; k <= 2048; k <<= 1) {
        for (int j = k >> 1; j > 0; j >>= 1) {
#pragma unroll
            for (int w = 0; w < 2; w++) {
                int i = tid + (w << 10);
                int l = i ^ j;
                if (l > i) {
                    unsigned long long a = keys[i], b = keys[l];
                    bool up = ((i & k) == 0);
                    bool sw = up ? (a < b) : (a > b);
                    if (sw) { keys[i] = b; keys[l] = a; }
                }
            }
            __syncthreads();
        }
    }

    // ---- decode top 1024 ----
    {
        unsigned long long kk = keys[tid];
        unsigned ub = (unsigned)(kk >> 32);
        unsigned id = 0xFFFFFFFFu - (unsigned)(kk & 0xFFFFFFFFull);
        float v = __uint_as_float(ub);
        val_s[tid] = v;
        idx_s[tid] = (int)id;
        ys_s[tid] = (float)(int)(id / 320u);
        xs_s[tid] = (float)(int)(id % 320u);
        keep_s[tid] = (v > 0.015f) ? 1 : 0;
    }
    __syncthreads();

    // ---- greedy NMS: IoU of equal 4x4 squares ----
    for (int i = 0; i < 1024; i++) {
        if (keep_s[i]) {
            int t = tid;
            if (t > i && keep_s[t]) {
                float ih = fmaxf(4.0f - fabsf(ys_s[t] - ys_s[i]), 0.f);
                float iw = fmaxf(4.0f - fabsf(xs_s[t] - xs_s[i]), 0.f);
                float inter = ih * iw;
                float iou = inter / (32.0f - inter);
                if (iou > 0.1f) keep_s[t] = 0;
            }
        }
        __syncthreads();
    }

    // ---- keep_top_k = 300 (cumsum over sorted keep) ----
    if (tid == 0) {
        int run = 0;
        for (int i = 0; i < 1024; i++) {
            if (keep_s[i]) { run++; if (run > 300) keep_s[i] = 0; }
        }
    }
    __syncthreads();

    // ---- zero-fill + scatter ----
    for (int i = tid; i < HWPX; i += 1024) { pn[i] = 0.f; pr[i] = 0.f; }
    __syncthreads();
    if (keep_s[tid]) {
        float v = val_s[tid];
        int id = idx_s[tid];
        if (id >= 0 && id < HWPX) {
            pn[id] = v;
            pr[id] = (v >= 0.015f) ? v : 0.f;
        }
    }
}

// ---------------------------------------------------------------------------
// host side
// ---------------------------------------------------------------------------
static void launch_conv(const _Float16* act, const _Float16* Bw, const float* bias,
                        _Float16* o16, float* o32, int Nimg, int H, int W,
                        int cin, int cout, int coutPad, int ks, int relu,
                        hipStream_t st) {
    int M = Nimg * H * W;
    int Mtiles = (M + 15) / 16;
    int Ngroups = (coutPad / 16 + 3) / 4;
    int waves = Mtiles * Ngroups;
    int blocks = (waves + 7) / 8;
    conv_wmma<<<blocks, 256, 0, st>>>(act, Bw, bias, o16, o32,
                                      Nimg, H, W, cin, cout, coutPad, ks, relu);
}

extern "C" void kernel_launch(void* const* d_in, const int* in_sizes, int n_in,
                              void* d_out, int out_size, void* d_ws, size_t ws_size,
                              hipStream_t stream) {
    const float* x = (const float*)d_in[0];
    const float* w[10]; const float* b[10];
    for (int i = 0; i < 10; i++) {
        w[i] = (const float*)d_in[1 + 2 * i];
        b[i] = (const float*)d_in[2 + 2 * i];
    }

    char* ws = (char*)d_ws;
    size_t off = 0;
    auto carve = [&](size_t bytes) -> char* {
        char* pp = ws + off;
        off = (off + bytes + 255) & ~(size_t)255;
        return pp;
    };

    // layer configs for WMMA convs 1..9: {cin, cout, coutPad, ks}
    const int cfg[10][4] = {
        {1, 64, 64, 3},      // conv0 (direct, unused slot for Bmat)
        {64, 64, 64, 3},     // conv1
        {64, 64, 64, 3},     // conv2
        {64, 64, 64, 3},     // conv3
        {64, 128, 128, 3},   // conv4
        {128, 128, 128, 3},  // conv5
        {128, 128, 128, 3},  // conv6
        {128, 128, 128, 3},  // conv7
        {128, 256, 256, 3},  // conv8
        {256, 65, 80, 1},    // conv9 (head, 1x1)
    };

    _Float16* wb[10] = {};
    for (int i = 1; i < 10; i++) {
        int K = cfg[i][3] * cfg[i][3] * cfg[i][0];
        wb[i] = (_Float16*)carve((size_t)K * cfg[i][2] * sizeof(_Float16));
    }
    const size_t ARENA = (size_t)8 * 240 * 320 * 64;      // 39,321,600 f16
    _Float16* arA = (_Float16*)carve(ARENA * sizeof(_Float16));
    _Float16* arB = (_Float16*)carve(ARENA * sizeof(_Float16));
    float* semi = (float*)carve((size_t)9600 * 65 * sizeof(float));
    (void)ws_size; (void)in_sizes; (void)n_in; (void)out_size;

    // weight reformats
    for (int i = 1; i < 10; i++) {
        int K = cfg[i][3] * cfg[i][3] * cfg[i][0];
        int total = K * cfg[i][2];
        reformat_w<<<(total + 255) / 256, 256, 0, stream>>>(
            w[i], wb[i], cfg[i][0], cfg[i][1], cfg[i][2], cfg[i][3]);
    }

    // conv0 direct: x -> arA (240x320,64)
    {
        int total = 8 * 240 * 320 * 64;
        conv0_direct<<<(total + 255) / 256, 256, 0, stream>>>(x, w[0], b[0], arA);
    }

    // backbone (WMMA implicit GEMM), ping-pong arenas
    launch_conv(arA, wb[1], b[1], arB, nullptr, 8, 240, 320, 64, 64, 64, 3, 1, stream);
    { int t = 8 * 120 * 160 * 64;
      maxpool2<<<(t + 255) / 256, 256, 0, stream>>>(arB, arA, 8, 240, 320, 64); }
    launch_conv(arA, wb[2], b[2], arB, nullptr, 8, 120, 160, 64, 64, 64, 3, 1, stream);
    launch_conv(arB, wb[3], b[3], arA, nullptr, 8, 120, 160, 64, 64, 64, 3, 1, stream);
    { int t = 8 * 60 * 80 * 64;
      maxpool2<<<(t + 255) / 256, 256, 0, stream>>>(arA, arB, 8, 120, 160, 64); }
    launch_conv(arB, wb[4], b[4], arA, nullptr, 8, 60, 80, 64, 128, 128, 3, 1, stream);
    launch_conv(arA, wb[5], b[5], arB, nullptr, 8, 60, 80, 128, 128, 128, 3, 1, stream);
    { int t = 8 * 30 * 40 * 128;
      maxpool2<<<(t + 255) / 256, 256, 0, stream>>>(arB, arA, 8, 60, 80, 128); }
    launch_conv(arA, wb[6], b[6], arB, nullptr, 8, 30, 40, 128, 128, 128, 3, 1, stream);
    launch_conv(arB, wb[7], b[7], arA, nullptr, 8, 30, 40, 128, 128, 128, 3, 1, stream);
    // head
    launch_conv(arA, wb[8], b[8], arB, nullptr, 8, 30, 40, 128, 256, 256, 3, 1, stream);
    launch_conv(arB, wb[9], b[9], nullptr, semi, 8, 30, 40, 256, 65, 80, 1, 0, stream);

    // softmax + pixel shuffle -> prob (d_out[0 .. 614400))
    float* prob = (float*)d_out;
    softmax_shuffle<<<(9600 + 255) / 256, 256, 0, stream>>>(semi, prob);

    // per-image NMS -> prob_nms, pred
    nms_kernel<<<8, 1024, 0, stream>>>(prob, prob + 614400, prob + 2 * 614400);
}
